// MultiHeadAttention_17205638988108
// MI455X (gfx1250) — compile-verified, matrix operand order checked by reading, not run on hardware
//
#include <hip/hip_runtime.h>
#include <stddef.h>

typedef unsigned short u16;
typedef unsigned int u32;
typedef __attribute__((ext_vector_type(16))) __bf16 v16bf;
typedef __attribute__((ext_vector_type(8)))  __bf16 v8bf;
typedef __attribute__((ext_vector_type(8)))  float  v8f;

#define LARGE_NEG (-1.0e9f)

// problem dims (compile-time: fold all address math to shifts/masks)
#define BATCH 4
#define SEQ   2048
#define DMODEL 1024
#define NHEAD 16
#define DEPTH 64
#define MROWS (BATCH * SEQ)     // 8192

__device__ __forceinline__ u16 f2bf(float f) {
    u32 u = __float_as_uint(f);
    u32 r = u + 0x7FFFu + ((u >> 16) & 1u);   // round-to-nearest-even
    return (u16)(r >> 16);
}

__device__ __forceinline__ v16bf cat8(v8bf lo, v8bf hi) {
    return __builtin_shufflevector(lo, hi, 0,1,2,3,4,5,6,7,8,9,10,11,12,13,14,15);
}

// ---------------------------------------------------------------- fp32 -> bf16
__global__ void cvt_f32_bf16(const float* __restrict__ in, u16* __restrict__ out, int n) {
    int i = blockIdx.x * blockDim.x + threadIdx.x;
    if (i < n) out[i] = f2bf(in[i]);
}

// ---------------------------------------------------------------- GEMM (bf16 in, f32 acc)
// C[8192,1024] = A @ W + bias ; 128x128 tile / workgroup, 8 waves of 32.
// Ping-pong LDS double buffering: global loads for tile k+1 staged in registers
// while computing tile k; one barrier per K-step.
// HS=1: store bf16 head-split [B,H,S,64]; HS=0: store f32 row-major.
template <int HS>
__global__ __launch_bounds__(256)
void gemm_bf16(const u16* __restrict__ A, const u16* __restrict__ W,
               const float* __restrict__ bias, u16* __restrict__ outB,
               float* __restrict__ outF) {
    __shared__ __align__(16) u16 As[2][128 * 40];   // [row][k] ld=40
    __shared__ __align__(16) u16 Bs[2][128 * 40];   // [n][k] (transposed) ld=40
    const int t    = threadIdx.x;
    const int lane = t & 31;
    const int wid  = t >> 5;
    const int lm   = lane & 15;
    const int hfl  = lane >> 4;
    const int koff = hfl * 8;
    const int bm = blockIdx.y * 128;
    const int bn = blockIdx.x * 128;
    const int wm = (wid & 3) * 32;    // 2 row tiles of 16
    const int wn = (wid >> 2) * 64;   // 4 col tiles of 16

    // per-thread staging coordinates
    const int arow = t >> 2;          // [0,64)
    const int akk  = (t & 3) * 8;
    const int wkk  = t >> 4;          // [0,16)
    const int wnb  = (t & 15) * 8;

    v8f acc[2][4] = {};
    v8bf va0, va1, vw0, vw1;

    // ---- prologue: stage K-tile 0
    va0 = *(const v8bf*)(A + (size_t)(bm + arow) * DMODEL + akk);
    va1 = *(const v8bf*)(A + (size_t)(bm + 64 + arow) * DMODEL + akk);
    vw0 = *(const v8bf*)(W + (size_t)(wkk) * DMODEL + bn + wnb);
    vw1 = *(const v8bf*)(W + (size_t)(16 + wkk) * DMODEL + bn + wnb);
    {
        *(v8bf*)(As[0] + arow * 40 + akk) = va0;
        *(v8bf*)(As[0] + (64 + arow) * 40 + akk) = va1;
        union { v8bf v; u16 s[8]; } U;
        U.v = vw0;
        #pragma unroll
        for (int j = 0; j < 8; ++j) Bs[0][(wnb + j) * 40 + wkk] = U.s[j];
        U.v = vw1;
        #pragma unroll
        for (int j = 0; j < 8; ++j) Bs[0][(wnb + j) * 40 + 16 + wkk] = U.s[j];
    }
    __syncthreads();

    const int NKT = DMODEL / 32;      // 32 K-steps
    for (int kt = 0; kt < NKT; ++kt) {
        const int cur = kt & 1;
        // ---- issue next tile's global loads (latency hidden behind WMMAs)
        if (kt + 1 < NKT) {
            const int k0 = (kt + 1) * 32;
            va0 = *(const v8bf*)(A + (size_t)(bm + arow) * DMODEL + k0 + akk);
            va1 = *(const v8bf*)(A + (size_t)(bm + 64 + arow) * DMODEL + k0 + akk);
            vw0 = *(const v8bf*)(W + (size_t)(k0 + wkk) * DMODEL + bn + wnb);
            vw1 = *(const v8bf*)(W + (size_t)(k0 + 16 + wkk) * DMODEL + bn + wnb);
        }
        // ---- compute from current buffer
        v16bf af[2], bf[4];
        #pragma unroll
        for (int i = 0; i < 2; ++i) {
            int m = wm + i * 16 + lm;
            v8bf lo = *(const v8bf*)(As[cur] + m * 40 + koff);
            v8bf hi = *(const v8bf*)(As[cur] + m * 40 + 16 + koff);
            af[i] = cat8(lo, hi);
        }
        #pragma unroll
        for (int j = 0; j < 4; ++j) {
            int n = wn + j * 16 + lm;
            v8bf lo = *(const v8bf*)(Bs[cur] + n * 40 + koff);
            v8bf hi = *(const v8bf*)(Bs[cur] + n * 40 + 16 + koff);
            bf[j] = cat8(lo, hi);
        }
        #pragma unroll
        for (int i = 0; i < 2; ++i)
            #pragma unroll
            for (int j = 0; j < 4; ++j)
                acc[i][j] = __builtin_amdgcn_wmma_f32_16x16x32_bf16(
                    false, af[i], false, bf[j], (short)0, acc[i][j], false, false);
        // ---- store staged registers into the other buffer
        if (kt + 1 < NKT) {
            const int nxt = cur ^ 1;
            *(v8bf*)(As[nxt] + arow * 40 + akk) = va0;
            *(v8bf*)(As[nxt] + (64 + arow) * 40 + akk) = va1;
            union { v8bf v; u16 s[8]; } U;
            U.v = vw0;
            #pragma unroll
            for (int j = 0; j < 8; ++j) Bs[nxt][(wnb + j) * 40 + wkk] = U.s[j];
            U.v = vw1;
            #pragma unroll
            for (int j = 0; j < 8; ++j) Bs[nxt][(wnb + j) * 40 + 16 + wkk] = U.s[j];
        }
        __syncthreads();
    }

    // ---- epilogue: C/D layout lane=(col), vgpr r -> row r+8*half
    #pragma unroll
    for (int i = 0; i < 2; ++i) {
        #pragma unroll
        for (int j = 0; j < 4; ++j) {
            #pragma unroll
            for (int r = 0; r < 8; ++r) {
                int m = bm + wm + i * 16 + r + hfl * 8;
                int n = bn + wn + j * 16 + lm;
                float val = acc[i][j][r] + bias[n];
                if (HS) {
                    int b = m >> 11, s = m & (SEQ - 1);      // SEQ = 2048
                    int h = n >> 6, d = n & (DEPTH - 1);
                    outB[(((size_t)(b * NHEAD + h)) * SEQ + s) * DEPTH + d] = f2bf(val);
                } else {
                    outF[(size_t)m * DMODEL + n] = val;
                }
            }
        }
    }
}

// ---------------------------------------------------------------- flash attention
// Qh/Kh/Vh: bf16 [B*H][S][64]. One workgroup: 128 q-rows of one (b,h).
// Wave w owns q rows [16w,16w+16) across full 128-key tile -> softmax stats per-wave.
#define PS_OFF 0            // Ps: 128 x 136  (aliases Qs; Q consumed into regs first)
#define QS_OFF 0            // Qs: 128 x 72
#define KS_OFF 17408        // Ks: 128 x 72
#define VT_OFF 26624        // Vt: 64 x 136 (transposed: [d][k])
#define ATTN_LDS_ELEMS (26624 + 64 * 136)

__global__ __launch_bounds__(256)
void attn_kernel(const u16* __restrict__ Qh, const u16* __restrict__ Kh,
                 const u16* __restrict__ Vh, const int* __restrict__ mask,
                 u16* __restrict__ ctx) {
    extern __shared__ __align__(16) u16 smem[];
    u16* Ps = smem + PS_OFF;
    u16* Qs = smem + QS_OFF;
    u16* Ks = smem + KS_OFF;
    u16* Vt = smem + VT_OFF;

    const int t    = threadIdx.x;
    const int lane = t & 31;
    const int wid  = t >> 5;
    const int lm   = lane & 15;
    const int hfl  = lane >> 4;
    const int koff = hfl * 8;
    const int bh = blockIdx.y;
    const int b  = bh >> 4;
    const int q0 = blockIdx.x * 128;
    const size_t base = (size_t)bh * SEQ * DEPTH;

    // per-thread tile-load coordinates (8 bf16 = 16B per chunk, 4 chunks)
    const int lrow = t >> 3;          // [0,32)
    const int lkk  = (t & 7) * 8;

    // ---- load Q tile 128x64 into LDS
    #pragma unroll
    for (int i = 0; i < 4; ++i) {
        int row = lrow + 32 * i;
        *(v8bf*)(Qs + row * 72 + lkk) =
            *(const v8bf*)(Qh + base + (size_t)(q0 + row) * DEPTH + lkk);
    }
    __syncthreads();

    // ---- Q fragments (2 k-chunks of 32 over depth) held in registers
    v16bf qf[2];
    {
        int m = wid * 16 + lm;
        #pragma unroll
        for (int c = 0; c < 2; ++c) {
            v8bf lo = *(const v8bf*)(Qs + m * 72 + c * 32 + koff);
            v8bf hi = *(const v8bf*)(Qs + m * 72 + c * 32 + 16 + koff);
            qf[c] = cat8(lo, hi);
        }
    }

    float rm[8], rl[8];
    #pragma unroll
    for (int r = 0; r < 8; ++r) { rm[r] = -INFINITY; rl[r] = 0.0f; }
    v8f o[4] = {};

    const int NKT = SEQ / 128;        // 16
    for (int kt = 0; kt < NKT; ++kt) {
        // ---- load K tile 128x64 (row major) and V tile transposed [d][k]
        #pragma unroll
        for (int i = 0; i < 4; ++i) {
            int row = lrow + 32 * i;
            *(v8bf*)(Ks + row * 72 + lkk) =
                *(const v8bf*)(Kh + base + (size_t)(kt * 128 + row) * DEPTH + lkk);
        }
        #pragma unroll
        for (int i = 0; i < 4; ++i) {
            int krow = lrow + 32 * i;
            union { v8bf v; u16 s[8]; } U;
            U.v = *(const v8bf*)(Vh + base + (size_t)(kt * 128 + krow) * DEPTH + lkk);
            #pragma unroll
            for (int j = 0; j < 8; ++j) Vt[(lkk + j) * 136 + krow] = U.s[j];
        }
        // ---- prefetch next tile into GL2 (gfx1250 global_prefetch_b8)
        if (kt + 1 < NKT) {
            __builtin_prefetch(Kh + base + (size_t)((kt + 1) * 128 + lrow * 4) * DEPTH + lkk, 0, 1);
            __builtin_prefetch(Vh + base + (size_t)((kt + 1) * 128 + lrow * 4) * DEPTH + lkk, 0, 1);
        }
        __syncthreads();

        // ---- S = Q K^T : 8 key tiles x 2 depth chunks
        v8f sacc[8] = {};
        #pragma unroll
        for (int c = 0; c < 2; ++c) {
            #pragma unroll
            for (int nt = 0; nt < 8; ++nt) {
                int n = nt * 16 + lm;
                v8bf lo = *(const v8bf*)(Ks + n * 72 + c * 32 + koff);
                v8bf hi = *(const v8bf*)(Ks + n * 72 + c * 32 + 16 + koff);
                sacc[nt] = __builtin_amdgcn_wmma_f32_16x16x32_bf16(
                    false, qf[c], false, cat8(lo, hi), (short)0, sacc[nt], false, false);
            }
        }

        // ---- scale + mask + per-lane row max
        float rmx[8];
        #pragma unroll
        for (int r = 0; r < 8; ++r) rmx[r] = -INFINITY;
        #pragma unroll
        for (int nt = 0; nt < 8; ++nt) {
            int kc = kt * 128 + nt * 16 + lm;
            float mz = (float)mask[b * SEQ + kc] * LARGE_NEG;
            #pragma unroll
            for (int r = 0; r < 8; ++r) {
                float v = sacc[nt][r] * 0.125f + mz;   // 1/sqrt(64)
                sacc[nt][r] = v;
                rmx[r] = fmaxf(rmx[r], v);
            }
        }
        // reduce max across the 16 lanes sharing each row-half
        #pragma unroll
        for (int r = 0; r < 8; ++r) {
            float v = rmx[r];
            v = fmaxf(v, __shfl_xor(v, 1, 16));
            v = fmaxf(v, __shfl_xor(v, 2, 16));
            v = fmaxf(v, __shfl_xor(v, 4, 16));
            v = fmaxf(v, __shfl_xor(v, 8, 16));
            rmx[r] = v;
        }
        float corr[8], rs[8];
        #pragma unroll
        for (int r = 0; r < 8; ++r) {
            float mnew = fmaxf(rm[r], rmx[r]);
            corr[r] = __expf(rm[r] - mnew);
            rm[r] = mnew;
            rs[r] = 0.0f;
        }
        #pragma unroll
        for (int nt = 0; nt < 8; ++nt)
            #pragma unroll
            for (int r = 0; r < 8; ++r) {
                float e = __expf(sacc[nt][r] - rm[r]);
                sacc[nt][r] = e;
                rs[r] += e;
            }
        #pragma unroll
        for (int r = 0; r < 8; ++r) {
            float v = rs[r];
            v += __shfl_xor(v, 1, 16);
            v += __shfl_xor(v, 2, 16);
            v += __shfl_xor(v, 4, 16);
            v += __shfl_xor(v, 8, 16);
            rl[r] = rl[r] * corr[r] + v;
        }
        #pragma unroll
        for (int dt = 0; dt < 4; ++dt)
            #pragma unroll
            for (int r = 0; r < 8; ++r) o[dt][r] *= corr[r];

        // ---- spill P (bf16) to this wave's private LDS strip, reload as A-fragments
        #pragma unroll
        for (int nt = 0; nt < 8; ++nt)
            #pragma unroll
            for (int r = 0; r < 8; ++r) {
                int q = wid * 16 + r + hfl * 8;
                int k = nt * 16 + lm;
                Ps[q * 136 + k] = f2bf(sacc[nt][r]);
            }

        // ---- O += P V : 4 k-chunks x 4 depth tiles
        #pragma unroll
        for (int c = 0; c < 4; ++c) {
            int m = wid * 16 + lm;
            v8bf plo = *(const v8bf*)(Ps + m * 136 + c * 32 + koff);
            v8bf phi = *(const v8bf*)(Ps + m * 136 + c * 32 + 16 + koff);
            v16bf pf = cat8(plo, phi);
            #pragma unroll
            for (int dt = 0; dt < 4; ++dt) {
                int d = dt * 16 + lm;
                v8bf vlo = *(const v8bf*)(Vt + d * 136 + c * 32 + koff);
                v8bf vhi = *(const v8bf*)(Vt + d * 136 + c * 32 + 16 + koff);
                o[dt] = __builtin_amdgcn_wmma_f32_16x16x32_bf16(
                    false, pf, false, cat8(vlo, vhi), (short)0, o[dt], false, false);
            }
        }
        __syncthreads();
    }

    // ---- normalize and store ctx (bf16, concat-head layout [B*S,1024])
    float invl[8];
    #pragma unroll
    for (int r = 0; r < 8; ++r) invl[r] = 1.0f / rl[r];
    const int hcol = (bh & (NHEAD - 1)) * DEPTH;
    #pragma unroll
    for (int dt = 0; dt < 4; ++dt)
        #pragma unroll
        for (int r = 0; r < 8; ++r) {
            int qrow = q0 + wid * 16 + r + hfl * 8;
            int d = dt * 16 + lm;
            ctx[((size_t)b * SEQ + qrow) * DMODEL + hcol + d] = f2bf(o[dt][r] * invl[r]);
        }
}

// ---------------------------------------------------------------- host launcher
extern "C" void kernel_launch(void* const* d_in, const int* in_sizes, int n_in,
                              void* d_out, int out_size, void* d_ws, size_t ws_size,
                              hipStream_t stream) {
    const float* v    = (const float*)d_in[0];
    const float* k    = (const float*)d_in[1];
    const float* q    = (const float*)d_in[2];
    const int*   mask = (const int*)d_in[3];
    const float* Wq = (const float*)d_in[4];
    const float* bq = (const float*)d_in[5];
    const float* Wk = (const float*)d_in[6];
    const float* bk = (const float*)d_in[7];
    const float* Wv = (const float*)d_in[8];
    const float* bv = (const float*)d_in[9];
    const float* Wo = (const float*)d_in[10];
    const float* bo = (const float*)d_in[11];

    const int M = MROWS, D = DMODEL;
    const size_t actB = (size_t)M * D * 2;
    const size_t wB   = (size_t)D * D * 2;

    char* ws = (char*)d_ws;
    size_t off = 0;
    auto alloc = [&](size_t bytes) { void* p = ws + off; off += (bytes + 255) & ~(size_t)255; return p; };
    u16* qb  = (u16*)alloc(actB);
    u16* kb  = (u16*)alloc(actB);
    u16* vb  = (u16*)alloc(actB);
    u16* Wqb = (u16*)alloc(wB);
    u16* Wkb = (u16*)alloc(wB);
    u16* Wvb = (u16*)alloc(wB);
    u16* Wob = (u16*)alloc(wB);
    u16* Qh  = (u16*)alloc(actB);
    u16* Kh  = (u16*)alloc(actB);
    u16* Vh  = (u16*)alloc(actB);
    u16* ctx = (u16*)alloc(actB);

    // fp32 -> bf16 conversions
    int nA = M * D, nW = D * D;
    cvt_f32_bf16<<<(nA + 255) / 256, 256, 0, stream>>>(q, qb, nA);
    cvt_f32_bf16<<<(nA + 255) / 256, 256, 0, stream>>>(k, kb, nA);
    cvt_f32_bf16<<<(nA + 255) / 256, 256, 0, stream>>>(v, vb, nA);
    cvt_f32_bf16<<<(nW + 255) / 256, 256, 0, stream>>>(Wq, Wqb, nW);
    cvt_f32_bf16<<<(nW + 255) / 256, 256, 0, stream>>>(Wk, Wkb, nW);
    cvt_f32_bf16<<<(nW + 255) / 256, 256, 0, stream>>>(Wv, Wvb, nW);
    cvt_f32_bf16<<<(nW + 255) / 256, 256, 0, stream>>>(Wo, Wob, nW);

    // Q/K/V projections -> head-split bf16
    dim3 gg(D / 128, M / 128);
    gemm_bf16<1><<<gg, 256, 0, stream>>>(qb, Wqb, bq, Qh, nullptr);
    gemm_bf16<1><<<gg, 256, 0, stream>>>(kb, Wkb, bk, Kh, nullptr);
    gemm_bf16<1><<<gg, 256, 0, stream>>>(vb, Wvb, bv, Vh, nullptr);

    // flash attention
    dim3 ga(SEQ / 128, BATCH * NHEAD);
    attn_kernel<<<ga, 256, ATTN_LDS_ELEMS * sizeof(u16), stream>>>(Qh, Kh, Vh, mask, ctx);

    // output projection -> fp32 d_out
    gemm_bf16<0><<<gg, 256, 0, stream>>>(ctx, Wob, bo, nullptr, (float*)d_out);
}
